// HMPNNLayer_15118284882724
// MI455X (gfx1250) — matrix operand order repaired and checked by reading, not compiled, not run on hardware
//
#include <hip/hip_runtime.h>
#include <hip/hip_bf16.h>

// ---------------------------------------------------------------------------
// HMPNN layer for MI455X (gfx1250, wave32, WMMA).
// Pipeline: BN-init -> zero agg -> WMMA GEMM1 (+sigmoid) -> scatter-add ->
//           WMMA GEMM2 (+sigmoid, K=256 concat) -> scatter-add -> finalize.
// ---------------------------------------------------------------------------

#define N_NODES 100000
#define N_EDGES 20000
#define NNZ_    640000
#define FDIM    128
#define BN_EPS  1e-5f

typedef __attribute__((ext_vector_type(16))) __bf16 v16bf;
typedef __attribute__((ext_vector_type(8)))  float  v8f;

__device__ __forceinline__ float sigmoidf_(float x) {
    return 1.0f / (1.0f + __expf(-x));
}

// ---------------------------------------------------------------------------
// out[i] = BN(x[i])   (pre-sigmoid accumulation base for x0 path)
// ---------------------------------------------------------------------------
__global__ __launch_bounds__(256) void bn_affine_kernel(
    const float* __restrict__ x,
    const float* __restrict__ gamma, const float* __restrict__ beta,
    const float* __restrict__ mean,  const float* __restrict__ var,
    float* __restrict__ out, int total)
{
    int i = blockIdx.x * 256 + threadIdx.x;
    if (i >= total) return;
    int f = i & (FDIM - 1);
    float sc = gamma[f] * rsqrtf(var[f] + BN_EPS);
    out[i] = (x[i] - mean[f]) * sc + beta[f];
}

__global__ __launch_bounds__(256) void zero_kernel(float* __restrict__ p, int total)
{
    int i = blockIdx.x * 256 + threadIdx.x;
    if (i < total) p[i] = 0.0f;
}

__global__ __launch_bounds__(256) void sigmoid_inplace_kernel(float* __restrict__ p, int total)
{
    int i = blockIdx.x * 256 + threadIdx.x;
    if (i < total) p[i] = sigmoidf_(p[i]);
}

// out[i] = sigmoid(BN(x[i]) + agg[i])   (x1 path)
__global__ __launch_bounds__(256) void bn_add_sigmoid_kernel(
    const float* __restrict__ x, const float* __restrict__ agg,
    const float* __restrict__ gamma, const float* __restrict__ beta,
    const float* __restrict__ mean,  const float* __restrict__ var,
    float* __restrict__ out, int total)
{
    int i = blockIdx.x * 256 + threadIdx.x;
    if (i >= total) return;
    int f = i & (FDIM - 1);
    float sc = gamma[f] * rsqrtf(var[f] + BN_EPS);
    out[i] = sigmoidf_((x[i] - mean[f]) * sc + beta[f] + agg[i]);
}

// ---------------------------------------------------------------------------
// dst[sidx[e]][f] += src[gidx[e]][f]  -- one thread per (nz, feature)
// ---------------------------------------------------------------------------
__global__ __launch_bounds__(256) void scatter_add_kernel(
    const float* __restrict__ src,
    const int*   __restrict__ gidx,  // gather row of src
    const int*   __restrict__ sidx,  // scatter row of dst
    float* __restrict__ dst, int nnz)
{
    long t = (long)blockIdx.x * 256 + threadIdx.x;
    int e = (int)(t >> 7);
    int f = (int)(t & (FDIM - 1));
    if (e >= nnz) return;
    int rs = gidx[e];
    int rd = sidx[e];
    float v = src[(size_t)rs * FDIM + f];
    __hip_atomic_fetch_add(&dst[(size_t)rd * FDIM + f], v,
                           __ATOMIC_RELAXED, __HIP_MEMORY_SCOPE_AGENT);
}

// ---------------------------------------------------------------------------
// WMMA GEMM: out[r][n] = sigmoid( sum_k A[r][k] * W[k][n] + bias[n] )
//   A is [nrows x KDIM] f32, formed as A0 (cols 0..127) ++ A1 (cols 128..255)
//   when KDIM == 256 (concat of x1 and node_msg_agg).  W is [KDIM x 128] f32.
// Block = 256 threads = 8 waves; each wave owns a 16-row tile, full 128 cols.
// Weights are transposed + converted to bf16 in LDS (Wt[n][k], k contiguous)
// so a B fragment is 32 contiguous bytes per lane.  A fragments are built
// directly from global f32 in contiguous k order, the same order B uses:
//   lane = (K-half << 4) | M ;  elems 0..15 -> K = kt*32 + 16*sel + i
// ---------------------------------------------------------------------------
template <int KDIM>
__global__ __launch_bounds__(256) void gemm_sigmoid_kernel(
    const float* __restrict__ A0,
    const float* __restrict__ A1,
    const float* __restrict__ W,     // [KDIM x 128] row-major f32
    const float* __restrict__ bias,  // [128]
    float* __restrict__ out,         // [nrows x 128]
    int nrows)
{
    __shared__ __attribute__((aligned(32))) __bf16 Wt[KDIM * FDIM]; // Wt[n*KDIM + k]

    // Cooperative load + transpose + f32->bf16 of the weight matrix.
    for (int idx = threadIdx.x; idx < KDIM * FDIM; idx += 256) {
        int k = idx >> 7;          // row of W
        int n = idx & (FDIM - 1);  // col of W
        Wt[n * KDIM + k] = (__bf16)W[idx];
    }
    __syncthreads();

    const int lane = threadIdx.x & 31;
    const int wave = threadIdx.x >> 5;
    const int sel  = lane >> 4;    // K-half / M-half selector
    const int nl   = lane & 15;    // row (A) / col (B,C) within tile

    const int rowBase = (blockIdx.x * 8 + wave) * 16;
    int arow = rowBase + nl;
    if (arow >= nrows) arow = nrows - 1;   // clamp loads; stores are masked

    v8f acc[8] = {};

    #pragma unroll
    for (int kt = 0; kt < KDIM / 32; ++kt) {
        const int c0 = kt * 32 + sel * 16;
        const float* rowp = A0;
        int cc = c0;
        if (KDIM == 256 && c0 >= 128) { rowp = A1; cc = c0 - 128; }
        const float* p = rowp + (size_t)arow * FDIM + cc;

        union { float4 v[4]; float f[16]; } u;
        u.v[0] = *(const float4*)(p);
        u.v[1] = *(const float4*)(p + 4);
        u.v[2] = *(const float4*)(p + 8);
        u.v[3] = *(const float4*)(p + 12);

        v16bf afrag;
        #pragma unroll
        for (int i = 0; i < 16; ++i) afrag[i] = (__bf16)u.f[i];

        #pragma unroll
        for (int nt = 0; nt < 8; ++nt) {
            const int ncol = nt * 16 + nl;
            // B fragment: 16 contiguous bf16 (K = kt*32 + 16*sel .. +15)
            v16bf bfrag = *(const v16bf*)(Wt + ncol * KDIM + kt * 32 + sel * 16);
            acc[nt] = __builtin_amdgcn_wmma_f32_16x16x32_bf16(
                false, afrag, false, bfrag, (short)0, acc[nt], false, false);
        }
    }

    // Epilogue: C/D layout -> lane L holds (M = 8*(L>>4)+j, N = L&15) per tile.
    #pragma unroll
    for (int nt = 0; nt < 8; ++nt) {
        const int ncol = nt * 16 + nl;
        const float bz = bias[ncol];
        #pragma unroll
        for (int j = 0; j < 8; ++j) {
            int mrow = rowBase + 8 * sel + j;
            if (mrow < nrows)
                out[(size_t)mrow * FDIM + ncol] = sigmoidf_(acc[nt][j] + bz);
        }
    }
}

// ---------------------------------------------------------------------------
extern "C" void kernel_launch(void* const* d_in, const int* in_sizes, int n_in,
                              void* d_out, int out_size, void* d_ws, size_t ws_size,
                              hipStream_t stream) {
    const float* x0       = (const float*)d_in[0];
    const float* x1       = (const float*)d_in[1];
    const int*   inc_node = (const int*)  d_in[2];
    const int*   inc_edge = (const int*)  d_in[3];
    const float* W_nh     = (const float*)d_in[4];
    const float* b_nh     = (const float*)d_in[5];
    const float* W_hn     = (const float*)d_in[6];
    const float* b_hn     = (const float*)d_in[7];
    const float* g0 = (const float*)d_in[8];
    const float* be0 = (const float*)d_in[9];
    const float* m0 = (const float*)d_in[10];
    const float* v0 = (const float*)d_in[11];
    const float* g1 = (const float*)d_in[12];
    const float* be1 = (const float*)d_in[13];
    const float* m1 = (const float*)d_in[14];
    const float* v1 = (const float*)d_in[15];

    float* out_x0 = (float*)d_out;                       // [N_NODES x F]
    float* out_x1 = out_x0 + (size_t)N_NODES * FDIM;     // [N_EDGES x F]

    float* ws_node_msg = (float*)d_ws;                            // [N x F]
    float* ws_agg      = ws_node_msg + (size_t)N_NODES * FDIM;    // [E x F]
    float* ws_he_msg   = ws_agg      + (size_t)N_EDGES * FDIM;    // [E x F]

    const int totN = N_NODES * FDIM;   // 12.8M
    const int totE = N_EDGES * FDIM;   // 2.56M

    // 1) d_out.x0 = BN0(x0)  (accumulation base, pre-sigmoid)
    bn_affine_kernel<<<(totN + 255) / 256, 256, 0, stream>>>(
        x0, g0, be0, m0, v0, out_x0, totN);

    // 2) agg = 0
    zero_kernel<<<(totE + 255) / 256, 256, 0, stream>>>(ws_agg, totE);

    // 3) node_msg = sigmoid(x0 @ W_nh + b_nh)   [WMMA bf16]
    gemm_sigmoid_kernel<128><<<(N_NODES + 127) / 128, 256, 0, stream>>>(
        x0, x0, W_nh, b_nh, ws_node_msg, N_NODES);

    // 4) agg[inc_edge[e]] += node_msg[inc_node[e]]
    scatter_add_kernel<<<((long)NNZ_ * FDIM + 255) / 256, 256, 0, stream>>>(
        ws_node_msg, inc_node, inc_edge, ws_agg, NNZ_);

    // 5) he_msg = sigmoid(cat(x1, agg) @ W_hn + b_hn)   [WMMA bf16, K=256]
    gemm_sigmoid_kernel<256><<<(N_EDGES + 127) / 128, 256, 0, stream>>>(
        x1, ws_agg, W_hn, b_hn, ws_he_msg, N_EDGES);

    // 6) d_out.x0[inc_node[e]] += he_msg[inc_edge[e]]
    scatter_add_kernel<<<((long)NNZ_ * FDIM + 255) / 256, 256, 0, stream>>>(
        ws_he_msg, inc_edge, inc_node, out_x0, NNZ_);

    // 7) x0_out = sigmoid(x0_out)
    sigmoid_inplace_kernel<<<(totN + 255) / 256, 256, 0, stream>>>(out_x0, totN);

    // 8) x1_out = sigmoid(BN1(x1) + agg)
    bn_add_sigmoid_kernel<<<(totE + 255) / 256, 256, 0, stream>>>(
        x1, ws_agg, g1, be1, m1, v1, out_x1, totE);
}
